// rlstm_19610820674251
// MI455X (gfx1250) — matrix operand order repaired and checked by reading, not compiled, not run on hardware
//
#include <hip/hip_runtime.h>

// ---------------- problem constants ----------------
#define BB   256      // batch (NUM_PROP)
#define TT   200      // timesteps
#define HH   512      // hidden
#define GG   2048     // 4*H gates
#define C0   128      // layer0 input channels
#define C1   512      // layer1 input channels
#define NCLS 40

// ---------------- WMMA types ----------------
typedef __bf16 bf16;
typedef bf16  v16bf __attribute__((ext_vector_type(16)));
typedef float v8f   __attribute__((ext_vector_type(8)));
typedef unsigned int u32x4 __attribute__((ext_vector_type(4)));

union Frag { v16bf v; u32x4 q[2]; };

__device__ __forceinline__ unsigned short f2bf_rne(float x) {
    unsigned u = __float_as_uint(x);
    u += 0x7FFFu + ((u >> 16) & 1u);
    return (unsigned short)(u >> 16);
}

__device__ __forceinline__ float sigf(float x) {
    return 1.0f / (1.0f + __expf(-x));
}

// ---------------- prep kernels ----------------
__global__ void f2bf_kernel(const float* __restrict__ src,
                            unsigned short* __restrict__ dst, int n) {
    int i = blockIdx.x * blockDim.x + threadIdx.x;
    if (i < n) dst[i] = f2bf_rne(src[i]);
}

// Wcat[n][k] = k < C ? Wih[n][k] : Whh[n][k-C]   (2048 x ktot, bf16)
__global__ void build_wcat_kernel(const float* __restrict__ Wih,
                                  const float* __restrict__ Whh,
                                  unsigned short* __restrict__ out,
                                  int C, int ktot) {
    int i = blockIdx.x * blockDim.x + threadIdx.x;
    int n = i / ktot, k = i - n * ktot;
    if (n < GG) {
        float v = (k < C) ? Wih[n * C + k] : Whh[n * HH + (k - C)];
        out[i] = f2bf_rne(v);
    }
}

__global__ void bias_kernel(const float* __restrict__ bih,
                            const float* __restrict__ bhh,
                            float* __restrict__ out) {
    int i = blockIdx.x * blockDim.x + threadIdx.x;
    if (i < GG) out[i] = bih[i] + bhh[i];
}

// ---------------- grid-wide barrier ----------------
__device__ __forceinline__ void grid_barrier(unsigned* cnt, unsigned target) {
    __syncthreads();
    if (threadIdx.x == 0) {
        __threadfence();
        __hip_atomic_fetch_add(cnt, 1u, __ATOMIC_ACQ_REL, __HIP_MEMORY_SCOPE_AGENT);
        while (__hip_atomic_load(cnt, __ATOMIC_ACQUIRE, __HIP_MEMORY_SCOPE_AGENT) < target) {
            __builtin_amdgcn_s_sleep(2);
        }
    }
    __syncthreads();
}

// ---------------- per-layer GEMM step (fully unrolled, immediate offsets) ----
// Wave tile 32x64: two A row-blocks (M and M+16) share each B fragment, so each
// B load pair feeds two consecutive WMMAs. All offsets are compile-time.
template <int C>
__device__ __forceinline__ void gemm_step(const unsigned short* __restrict__ xbase,
                                          const unsigned short* __restrict__ hbase,
                                          const unsigned short* __restrict__ bbase,
                                          v8f acc[2][4]) {
    constexpr int KT   = C + HH;       // 640 or 1024
    constexpr int CCH  = C >> 5;       // x-fed chunks
    constexpr int NKC  = KT >> 5;      // total chunks of K=32
    constexpr int XR16 = 16 * TT * C;  // +16 batch rows in x panel (elements)
    constexpr int HR16 = 16 * HH;      // +16 batch rows in h panel (elements)
    #pragma unroll
    for (int kc = 0; kc < NKC; ++kc) {
        constexpr int z = 0; (void)z;
        const bool fromX = (kc < CCH);                  // compile-time per iter
        const unsigned short* ap =
            fromX ? (xbase + kc * 32) : (hbase + (kc - CCH) * 32);
        const int rstep = fromX ? XR16 : HR16;          // compile-time per iter
        Frag a0, a1;
        a0.q[0] = *(const u32x4*)(ap);
        a0.q[1] = *(const u32x4*)(ap + 16);
        a1.q[0] = *(const u32x4*)(ap + rstep);
        a1.q[1] = *(const u32x4*)(ap + rstep + 16);
        #pragma unroll
        for (int ct = 0; ct < 4; ++ct) {
            Frag b;
            const unsigned short* bp = bbase + ct * (16 * KT) + kc * 32;
            b.q[0] = *(const u32x4*)(bp);
            b.q[1] = *(const u32x4*)(bp + 16);
            acc[0][ct] = __builtin_amdgcn_wmma_f32_16x16x32_bf16(
                false, a0.v, false, b.v, (short)0, acc[0][ct], false, false);
            acc[1][ct] = __builtin_amdgcn_wmma_f32_16x16x32_bf16(
                false, a1.v, false, b.v, (short)0, acc[1][ct], false, false);
        }
    }
}

// ---------------- persistent 2-layer LSTM ----------------
// Grid: 32 WGs x 256 threads (8 wave32). WG tile of gates: 64 rows x 256 cols.
// Waves arranged 2 (row blocks of 32) x 4 (col blocks of 64).
__global__ void __launch_bounds__(256, 1)
lstm_persistent(const unsigned short* __restrict__ xbf0,   // [B,T,128] bf16
                unsigned short* __restrict__ hs0,          // [B,T,512] bf16 (layer0 out)
                const unsigned short* __restrict__ wcat0,  // [2048,640] bf16
                const unsigned short* __restrict__ wcat1,  // [2048,1024] bf16
                const float* __restrict__ bias0,           // [2048]
                const float* __restrict__ bias1,
                float* __restrict__ gbuf,                  // [256,2048] f32
                unsigned short* __restrict__ hbuf,         // [256,512] bf16
                float* __restrict__ cbuf,                  // [256,512] f32
                float* __restrict__ basefeat,              // [256,512] f32
                unsigned* __restrict__ counter) {
    const int tid  = threadIdx.x;
    const int lane = tid & 31;
    const int wave = tid >> 5;
    const int wg   = blockIdx.x;
    const int NWG  = gridDim.x;
    const int gid  = wg * blockDim.x + tid;
    const int gsz  = NWG * blockDim.x;

    // gate-matrix tiling
    const int wgRow = wg & 3;          // 4 row tiles of 64
    const int wgCol = wg >> 2;         // 8 col tiles of 256
    const int r0 = wgRow * 64 + (wave & 1) * 32;    // wave's 32 rows (batch idx)
    const int c0 = wgCol * 256 + (wave >> 1) * 64;  // wave's 64 gate cols
    const int mrow = r0 + (lane & 15);              // A-frag row for this lane
    const int alo  = (lane < 16) ? 0 : 8;           // A K-offset (16-bit A layout)
    const int blo  = (lane < 16) ? 0 : 8;           // B K-offset (16-bit B layout)
    const int bn   = lane & 15;                     // B column within 16-tile

    unsigned ep = 0;

    for (int layer = 0; layer < 2; ++layer) {
        const unsigned short* wcat = layer ? wcat1 : wcat0;
        const float* bias          = layer ? bias1 : bias0;
        const unsigned short* xsrc = layer ? hs0   : xbf0;
        const int C    = layer ? C1 : C0;
        const int ktot = C + HH;

        // zero h, c
        for (int idx = gid; idx < BB * HH; idx += gsz) {
            cbuf[idx] = 0.0f;
            hbuf[idx] = 0;
        }
        ++ep; grid_barrier(counter, ep * NWG);

        // lane-variant base pointers (three 64-bit pointers live in k-loop)
        const unsigned short* hbase = hbuf + mrow * HH + alo;
        const unsigned short* bbase = wcat + (size_t)(c0 + bn) * ktot + blo;

        for (int t = 0; t < TT; ++t) {
            // ---- gates[b,g] = [x_t | h] @ Wcat^T  (bf16 WMMA, f32 acc) ----
            v8f acc[2][4];
            #pragma unroll
            for (int mi = 0; mi < 2; ++mi)
                #pragma unroll
                for (int ct = 0; ct < 4; ++ct) acc[mi][ct] = 0.0f;

            const unsigned short* xbase = xsrc + ((size_t)mrow * TT + t) * C + alo;

            if (layer == 0) gemm_step<C0>(xbase, hbase, bbase, acc);
            else            gemm_step<C1>(xbase, hbase, bbase, acc);

            // store D tiles (f32 C/D layout: lanes<16 -> M=r, lanes>=16 -> M=r+8)
            #pragma unroll
            for (int mi = 0; mi < 2; ++mi) {
                #pragma unroll
                for (int ct = 0; ct < 4; ++ct) {
                    const int gcol = c0 + ct * 16 + bn;
                    #pragma unroll
                    for (int r = 0; r < 8; ++r) {
                        const int grow = r0 + mi * 16 + ((lane < 16) ? r : r + 8);
                        gbuf[grow * GG + gcol] = acc[mi][ct][r];
                    }
                }
            }
            ++ep; grid_barrier(counter, ep * NWG);

            // ---- LSTM cell elementwise: 256*512 cells ----
            for (int idx = gid; idx < BB * HH; idx += gsz) {
                const int b = idx >> 9;
                const int j = idx & (HH - 1);
                const float* gr = gbuf + b * GG;
                const float ig = sigf(gr[j]            + bias[j]);
                const float fg = sigf(gr[HH + j]       + bias[HH + j]);
                const float gg = tanhf(gr[2 * HH + j]  + bias[2 * HH + j]);
                const float og = sigf(gr[3 * HH + j]   + bias[3 * HH + j]);
                const float c  = fg * cbuf[idx] + ig * gg;
                const float h  = og * tanhf(c);
                cbuf[idx] = c;
                hbuf[idx] = f2bf_rne(h);
                if (layer == 0)
                    hs0[((size_t)b * TT + t) * HH + j] = f2bf_rne(h);
                else if (t == TT - 1)
                    basefeat[idx] = h;
            }
            ++ep; grid_barrier(counter, ep * NWG);
        }
    }
}

// ---------------- classification / bbox head ----------------
__global__ void head_kernel(const float* __restrict__ basefeat,
                            const float* __restrict__ clsW,
                            const float* __restrict__ clsb,
                            const float* __restrict__ bboxW,
                            const float* __restrict__ bboxb,
                            float* __restrict__ out) {
    int gid = blockIdx.x * blockDim.x + threadIdx.x;
    if (gid < BB * (NCLS + 2)) {
        const int b = gid / (NCLS + 2);
        const int o = gid - b * (NCLS + 2);
        const float* w = (o < NCLS) ? (clsW + o * HH) : (bboxW + (o - NCLS) * HH);
        const float  bi = (o < NCLS) ? clsb[o] : bboxb[o - NCLS];
        const float* f = basefeat + b * HH;
        float s = 0.0f;
        #pragma unroll 4
        for (int k = 0; k < HH; ++k) s = fmaf(f[k], w[k], s);
        s += bi;
        if (o < NCLS) out[b * NCLS + o] = s;
        else          out[BB * NCLS + b * 2 + (o - NCLS)] = s;
    }
    if (gid == 0) {
        out[BB * NCLS + BB * 2]     = 0.0f;  // cls_loss
        out[BB * NCLS + BB * 2 + 1] = 0.0f;  // bbox_loss
    }
}

// ---------------- launch ----------------
extern "C" void kernel_launch(void* const* d_in, const int* in_sizes, int n_in,
                              void* d_out, int out_size, void* d_ws, size_t ws_size,
                              hipStream_t stream) {
    const float* proposals = (const float*)d_in[2];   // [256,200,128]
    const float* Wih0 = (const float*)d_in[4];
    const float* Whh0 = (const float*)d_in[5];
    const float* bih0 = (const float*)d_in[6];
    const float* bhh0 = (const float*)d_in[7];
    const float* Wih1 = (const float*)d_in[8];
    const float* Whh1 = (const float*)d_in[9];
    const float* bih1 = (const float*)d_in[10];
    const float* bhh1 = (const float*)d_in[11];
    const float* clsW = (const float*)d_in[12];
    const float* clsb = (const float*)d_in[13];
    const float* bbW  = (const float*)d_in[14];
    const float* bbb  = (const float*)d_in[15];
    float* out = (float*)d_out;

    // workspace carve-out (256B aligned)
    char* p = (char*)d_ws;
    auto carve = [&](size_t bytes) -> void* {
        void* r = (void*)p;
        p += (bytes + 255) & ~(size_t)255;
        return r;
    };
    unsigned*       counter = (unsigned*)      carve(256);
    unsigned short* xbf0    = (unsigned short*)carve((size_t)BB * TT * C0 * 2);
    unsigned short* hs0     = (unsigned short*)carve((size_t)BB * TT * HH * 2);
    unsigned short* wcat0   = (unsigned short*)carve((size_t)GG * (C0 + HH) * 2);
    unsigned short* wcat1   = (unsigned short*)carve((size_t)GG * (C1 + HH) * 2);
    float*          bias0   = (float*)         carve((size_t)GG * 4);
    float*          bias1   = (float*)         carve((size_t)GG * 4);
    float*          gbuf    = (float*)         carve((size_t)BB * GG * 4);
    unsigned short* hbuf    = (unsigned short*)carve((size_t)BB * HH * 2);
    float*          cbuf    = (float*)         carve((size_t)BB * HH * 4);
    float*          bfeat   = (float*)         carve((size_t)BB * HH * 4);

    // reset grid-barrier counter every call (graph-capture safe)
    hipMemsetAsync(counter, 0, 256, stream);

    // prep: bf16 conversions + weight concatenation
    {
        int n = BB * TT * C0;
        f2bf_kernel<<<(n + 255) / 256, 256, 0, stream>>>(proposals, xbf0, n);
    }
    {
        int n = GG * (C0 + HH);
        build_wcat_kernel<<<(n + 255) / 256, 256, 0, stream>>>(Wih0, Whh0, wcat0, C0, C0 + HH);
    }
    {
        int n = GG * (C1 + HH);
        build_wcat_kernel<<<(n + 255) / 256, 256, 0, stream>>>(Wih1, Whh1, wcat1, C1, C1 + HH);
    }
    bias_kernel<<<GG / 256, 256, 0, stream>>>(bih0, bhh0, bias0);
    bias_kernel<<<GG / 256, 256, 0, stream>>>(bih1, bhh1, bias1);

    // persistent recurrent kernel: 32 WGs (must be co-resident for grid barrier)
    lstm_persistent<<<32, 256, 0, stream>>>(xbf0, hs0, wcat0, wcat1, bias0, bias1,
                                            gbuf, hbuf, cbuf, bfeat, counter);

    // head: 256 x 42 outputs + 2 zero losses
    head_kernel<<<(BB * (NCLS + 2) + 255) / 256, 256, 0, stream>>>(
        bfeat, clsW, clsb, bbW, bbb, out);
}
